// MultiHeadAttention_18683107737900
// MI455X (gfx1250) — matrix operand order repaired and checked by reading, not compiled, not run on hardware
//
#include <hip/hip_runtime.h>
#include <hip/hip_bf16.h>

// ---------------- CDNA5 WMMA types ----------------
typedef __bf16 bf16_t;
typedef __attribute__((ext_vector_type(16))) __bf16 v16bf;
typedef __attribute__((ext_vector_type(8)))  __bf16 bf16x8;
typedef __attribute__((ext_vector_type(8)))  float  v8f;
typedef __attribute__((ext_vector_type(4)))  float  f32x4;

static __device__ __forceinline__ v8f vzero8() {
    v8f z;
#pragma unroll
    for (int i = 0; i < 8; ++i) z[i] = 0.0f;
    return z;
}

static __device__ __forceinline__ bf16x8 ldb8(const unsigned short* p) {
    return *(const bf16x8*)p;
}

static __device__ __forceinline__ v16bf cat8(bf16x8 lo, bf16x8 hi) {
    return __builtin_shufflevector(lo, hi, 0,1,2,3,4,5,6,7,8,9,10,11,12,13,14,15);
}

static __device__ __forceinline__ v8f wmma_bf16(v16bf a, v16bf b, v8f c) {
    // D = A(16x32 bf16) * B(32x16 bf16) + C(16x16 f32)
    return __builtin_amdgcn_wmma_f32_16x16x32_bf16(
        /*neg_a=*/false, a, /*neg_b=*/false, b,
        /*c_mod=*/(short)0, c, /*reuse_a=*/false, /*reuse_b=*/false);
}

// round-to-nearest-even f32 -> bf16 (bit pattern as u16)
static __device__ __forceinline__ unsigned short f2bf(float f) {
    unsigned int u = __float_as_uint(f);
    u += 0x7FFFu + ((u >> 16) & 1u);
    return (unsigned short)(u >> 16);
}

#define NEG_INF_F (-1e30f)

// ---------------- problem constants ----------------
#define BB 4
#define SS 1024
#define DD 1024
#define HH 16
#define HD 64
#define MTOT (BB*SS)            // 4096
#define NEL  ((size_t)MTOT*DD)  // 4M elements
#define WN   ((size_t)DD*DD)    // 1M elements

// ---------------- f32 -> bf16 convert ----------------
__global__ void __launch_bounds__(256) cvt_f32_bf16(const float* __restrict__ in,
                                                    unsigned short* __restrict__ out,
                                                    int n) {
    for (int i = blockIdx.x * blockDim.x + threadIdx.x; i < n; i += gridDim.x * blockDim.x)
        out[i] = f2bf(in[i]);
}

// ---------------- WMMA GEMM:  Y = X @ W^T + bias  ----------------
// A: [M,K] bf16 row-major.  W: [N,K] bf16 row-major (so B = W^T).
// MODE 0: f32 out [M,N]  (final projection, writes d_out)
// MODE 1: bf16 out [M,N] scaled ((acc+bias)*scale)  (Q/K projections)
// MODE 2: bf16 out transposed per-head: Vt[b][h][nl][s]  (V projection)
template<int MODE>
__global__ void __launch_bounds__(256) gemm_wmma(const unsigned short* __restrict__ A,
                                                 const unsigned short* __restrict__ W,
                                                 const float* __restrict__ bias,
                                                 void* __restrict__ out,
                                                 int M, int N, int K, float scale) {
    const int lane = threadIdx.x & 31;
    const int wave = threadIdx.x >> 5;
    const int ln = lane & 15;
    const int hl = lane >> 4;
    const int m0 = blockIdx.x * 128 + (wave >> 1) * 32;
    const int n0 = blockIdx.y * 128 + (wave & 1) * 64;

    v8f acc[2][4];
#pragma unroll
    for (int mi = 0; mi < 2; ++mi)
#pragma unroll
        for (int ni = 0; ni < 4; ++ni) acc[mi][ni] = vzero8();

    for (int kk = 0; kk < K; kk += 32) {
        v16bf afrag[2];
#pragma unroll
        for (int mi = 0; mi < 2; ++mi) {
            const unsigned short* ap = A + (size_t)(m0 + mi * 16 + ln) * K + kk;
            afrag[mi] = cat8(ldb8(ap + hl * 8), ldb8(ap + 16 + hl * 8));
        }
        v16bf bfrag[4];
#pragma unroll
        for (int ni = 0; ni < 4; ++ni) {
            const unsigned short* wp = W + (size_t)(n0 + ni * 16 + ln) * K + kk;
            bfrag[ni] = cat8(ldb8(wp + hl * 8), ldb8(wp + 16 + hl * 8));
        }
        if (kk + 32 < K) { // CDNA5 global_prefetch_b8 for the next k-slab
            __builtin_prefetch(A + (size_t)(m0 + ln) * K + kk + 32, 0, 1);
            __builtin_prefetch(W + (size_t)(n0 + ln) * K + kk + 32, 0, 1);
        }
#pragma unroll
        for (int mi = 0; mi < 2; ++mi)
#pragma unroll
            for (int ni = 0; ni < 4; ++ni)
                acc[mi][ni] = wmma_bf16(afrag[mi], bfrag[ni], acc[mi][ni]);
    }

#pragma unroll
    for (int ni = 0; ni < 4; ++ni) {
        const int n = n0 + ni * 16 + ln;
        const float bn = bias[n];
#pragma unroll
        for (int mi = 0; mi < 2; ++mi) {
#pragma unroll
            for (int r = 0; r < 8; ++r) {
                const int m = m0 + mi * 16 + r + 8 * hl;
                const float v = (acc[mi][ni][r] + bn) * scale;
                if (MODE == 0) {
                    ((float*)out)[(size_t)m * N + n] = v;
                } else if (MODE == 1) {
                    ((unsigned short*)out)[(size_t)m * N + n] = f2bf(v);
                } else { // MODE 2: Vt[((b*H+head)*HD+nl)*S + s]
                    const int b = m >> 10, s = m & 1023;
                    const int head = n >> 6, nl = n & 63;
                    ((unsigned short*)out)[(size_t)((b * HH + head) * HD + nl) * SS + s] = f2bf(v);
                }
            }
        }
    }
}

// masked 16x16 score tile for columns kcol (per lane), rows q0..q0+15
static __device__ __forceinline__ v8f score_tile(const unsigned short* __restrict__ KP_bh,
                                                 v16bf aq0, v16bf aq1,
                                                 const unsigned char* __restrict__ km,
                                                 const unsigned char* const* amrow,
                                                 int kcol, int hl) {
    const unsigned short* kp = KP_bh + (size_t)kcol * DD;
    v8f c = vzero8();
    v16bf bk0 = cat8(ldb8(kp + hl * 8), ldb8(kp + 16 + hl * 8));
    c = wmma_bf16(aq0, bk0, c);
    v16bf bk1 = cat8(ldb8(kp + 32 + hl * 8), ldb8(kp + 48 + hl * 8));
    c = wmma_bf16(aq1, bk1, c);
    const unsigned int kmv = (unsigned int)km[kcol];
#pragma unroll
    for (int r = 0; r < 8; ++r) {
        const unsigned int amv = (unsigned int)amrow[r][kcol];
        c[r] = ((kmv & amv) != 0u) ? c[r] : NEG_INF_F;
    }
    return c;
}

// ---------------- fused attention ----------------
// One workgroup = (b, 16 q-rows), 16 waves = 16 heads (512 threads).
// Per wave: three sweeps over K with lane-local reductions (cross-lane
// ds_bpermute only twice, outside the loops).  Sweep 3 stages the 16x32
// probability block of every head in LDS, then the whole workgroup writes
// aw[q][k][h] fully coalesced (complete 64B runs / full cachelines), and
// each wave WMMA-accumulates O = P @ V from its LDS bf16 tile (transposed V).
__global__ void __launch_bounds__(512) attn_fused(const unsigned short* __restrict__ QP,
                                                  const unsigned short* __restrict__ KP,
                                                  const unsigned short* __restrict__ VT,
                                                  const unsigned char* __restrict__ key_mask,
                                                  const unsigned char* __restrict__ attn_mask,
                                                  float* __restrict__ aw,
                                                  unsigned short* __restrict__ ctx) {
    const int tid = threadIdx.x;
    const int lane = tid & 31;
    const int head = tid >> 5;     // wave id == head
    const int ln = lane & 15;
    const int hl = lane >> 4;
    const int q0 = blockIdx.x * 16;
    const int b = blockIdx.y;

    // Q fragments for this 16-row block: A is 16x64 -> two 16x32 fragments
    v16bf aq0, aq1;
    {
        const unsigned short* qp = QP + ((size_t)(b * SS + q0 + ln) * DD + head * HD);
        aq0 = cat8(ldb8(qp + hl * 8), ldb8(qp + 16 + hl * 8));
        aq1 = cat8(ldb8(qp + 32 + hl * 8), ldb8(qp + 48 + hl * 8));
    }

    const unsigned short* KP_bh = KP + (size_t)(b * SS) * DD + head * HD;
    const unsigned char* km = key_mask + (size_t)b * SS;
    const unsigned char* amrow[8];
#pragma unroll
    for (int r = 0; r < 8; ++r)
        amrow[r] = attn_mask + (size_t)(q0 + r + 8 * hl) * SS;

    // -------- sweep 1: lane-local column max --------
    float vmax[8];
#pragma unroll
    for (int r = 0; r < 8; ++r) vmax[r] = -3.0e38f;
    for (int kt = 0; kt < SS / 16; ++kt) {
        v8f sv = score_tile(KP_bh, aq0, aq1, km, amrow, kt * 16 + ln, hl);
#pragma unroll
        for (int r = 0; r < 8; ++r) vmax[r] = fmaxf(vmax[r], sv[r]);
    }
#pragma unroll
    for (int r = 0; r < 8; ++r) {
#pragma unroll
        for (int off = 8; off > 0; off >>= 1)
            vmax[r] = fmaxf(vmax[r], __shfl_xor(vmax[r], off, 32));
    }

    // -------- sweep 2: lane-local exp-sum --------
    float psum[8];
#pragma unroll
    for (int r = 0; r < 8; ++r) psum[r] = 0.0f;
    for (int kt = 0; kt < SS / 16; ++kt) {
        v8f sv = score_tile(KP_bh, aq0, aq1, km, amrow, kt * 16 + ln, hl);
#pragma unroll
        for (int r = 0; r < 8; ++r) psum[r] += __expf(sv[r] - vmax[r]);
    }
    float linv[8];
#pragma unroll
    for (int r = 0; r < 8; ++r) {
#pragma unroll
        for (int off = 8; off > 0; off >>= 1)
            psum[r] += __shfl_xor(psum[r], off, 32);
        linv[r] = 1.0f / psum[r];
    }

    // -------- sweep 3: stage probs in LDS, coalesced aw write, O = P @ V ----
    v8f oacc[4];
#pragma unroll
    for (int ni = 0; ni < 4; ++ni) oacc[ni] = vzero8();

    // f32 staging [16 q][32 k][16 h + 1 pad]  (pad 17 -> conflict-free)
    __shared__ float Pf[16 * 32 * 17];                 // ~34 KB
    // bf16 A-tiles, one 16x32 tile per head
    __shared__ __align__(16) unsigned short Pb[16][16 * 32];   // 16 KB

    for (int kt2 = 0; kt2 < SS / 32; ++kt2) {
#pragma unroll
        for (int t = 0; t < 2; ++t) {
            const int kloc = t * 16 + ln;              // 0..31 within the k-block
            v8f sv = score_tile(KP_bh, aq0, aq1, km, amrow, kt2 * 32 + kloc, hl);
#pragma unroll
            for (int r = 0; r < 8; ++r) {
                const int qr = r + 8 * hl;
                const float p = __expf(sv[r] - vmax[r]) * linv[r];
                Pf[(qr * 32 + kloc) * 17 + head] = p;
                Pb[head][qr * 32 + kloc] = f2bf(p);
            }
        }
        __syncthreads();

        // coalesced aw writer: thread (wq = tid>>5, wk = lane) writes the 16
        // head-values of one (q,k) cell as four 16B stores (contiguous 64B).
        {
            const int wq = tid >> 5;      // 0..15
            const int wk = lane;          // 0..31
            const float* src = &Pf[(wq * 32 + wk) * 17];
            float* dst = aw + ((size_t)(b * SS + q0 + wq) * SS + kt2 * 32 + wk) * HH;
#pragma unroll
            for (int i = 0; i < 4; ++i) {
                f32x4 v4;
#pragma unroll
                for (int j = 0; j < 4; ++j) v4[j] = src[i * 4 + j];
                ((f32x4*)dst)[i] = v4;
            }
        }

        // P tile (16x32) as A fragment from LDS; B = transposed V rows
        v16bf ap = cat8(ldb8(&Pb[head][ln * 32 + hl * 8]),
                        ldb8(&Pb[head][ln * 32 + 16 + hl * 8]));
        const int kbase = kt2 * 32;
#pragma unroll
        for (int ni = 0; ni < 4; ++ni) {
            const unsigned short* vp =
                VT + ((size_t)((b * HH + head) * HD + ni * 16 + ln)) * SS + kbase;
            v16bf bv = cat8(ldb8(vp + hl * 8), ldb8(vp + 16 + hl * 8));
            oacc[ni] = wmma_bf16(ap, bv, oacc[ni]);
        }
        __syncthreads();
    }

    // ctx[b][q][head*64 + n] as bf16 (A operand of the final GEMM)
#pragma unroll
    for (int ni = 0; ni < 4; ++ni) {
#pragma unroll
        for (int r = 0; r < 8; ++r) {
            const int q = q0 + r + 8 * hl;
            ctx[(size_t)(b * SS + q) * DD + head * HD + ni * 16 + ln] = f2bf(oacc[ni][r]);
        }
    }
}

// ---------------- host-side orchestration ----------------
extern "C" void kernel_launch(void* const* d_in, const int* in_sizes, int n_in,
                              void* d_out, int out_size, void* d_ws, size_t ws_size,
                              hipStream_t stream) {
    (void)in_sizes; (void)n_in; (void)out_size; (void)ws_size;
    const float* q_in  = (const float*)d_in[0];
    const float* k_in  = (const float*)d_in[1];
    const float* v_in  = (const float*)d_in[2];
    const unsigned char* key_mask  = (const unsigned char*)d_in[3];
    const unsigned char* attn_mask = (const unsigned char*)d_in[4];
    const float* Wq = (const float*)d_in[5];  const float* bq = (const float*)d_in[6];
    const float* Wk = (const float*)d_in[7];  const float* bk = (const float*)d_in[8];
    const float* Wv = (const float*)d_in[9];  const float* bv = (const float*)d_in[10];
    const float* Wo = (const float*)d_in[11]; const float* bo = (const float*)d_in[12];

    float* out = (float*)d_out;                 // [4096,1024] f32
    float* aw  = (float*)d_out + NEL;           // [4,1024,1024,16] f32

    // workspace layout (bf16 elements): 7*NEL + 4*WN shorts = 64 MB
    unsigned short* ws = (unsigned short*)d_ws;
    unsigned short* XQ  = ws;                 // NEL
    unsigned short* XK  = XQ  + NEL;
    unsigned short* XV  = XK  + NEL;
    unsigned short* WQb = XV  + NEL;          // WN each
    unsigned short* WKb = WQb + WN;
    unsigned short* WVb = WKb + WN;
    unsigned short* WOb = WVb + WN;
    unsigned short* QP  = WOb + WN;           // NEL each
    unsigned short* KP  = QP  + NEL;
    unsigned short* VT  = KP  + NEL;
    unsigned short* CTX = VT  + NEL;

    const int cvt_blocks = 2048, cvt_threads = 256;
    cvt_f32_bf16<<<cvt_blocks, cvt_threads, 0, stream>>>(q_in, XQ, (int)NEL);
    cvt_f32_bf16<<<cvt_blocks, cvt_threads, 0, stream>>>(k_in, XK, (int)NEL);
    cvt_f32_bf16<<<cvt_blocks, cvt_threads, 0, stream>>>(v_in, XV, (int)NEL);
    cvt_f32_bf16<<<cvt_blocks, cvt_threads, 0, stream>>>(Wq, WQb, (int)WN);
    cvt_f32_bf16<<<cvt_blocks, cvt_threads, 0, stream>>>(Wk, WKb, (int)WN);
    cvt_f32_bf16<<<cvt_blocks, cvt_threads, 0, stream>>>(Wv, WVb, (int)WN);
    cvt_f32_bf16<<<cvt_blocks, cvt_threads, 0, stream>>>(Wo, WOb, (int)WN);

    const dim3 ggrid(MTOT / 128, DD / 128);   // (32, 8)
    const float scaling = 0.125f;             // hd^-0.5 = 64^-0.5
    gemm_wmma<1><<<ggrid, 256, 0, stream>>>(XQ, WQb, bq, (void*)QP, MTOT, DD, DD, scaling);
    gemm_wmma<1><<<ggrid, 256, 0, stream>>>(XK, WKb, bk, (void*)KP, MTOT, DD, DD, 1.0f);
    gemm_wmma<2><<<ggrid, 256, 0, stream>>>(XV, WVb, bv, (void*)VT, MTOT, DD, DD, 1.0f);

    const dim3 agrid(SS / 16, BB);            // (64, 4), 512 threads = 16 waves
    attn_fused<<<agrid, 512, 0, stream>>>(QP, KP, VT, key_mask, attn_mask, aw, CTX);

    gemm_wmma<0><<<ggrid, 256, 0, stream>>>(CTX, WOb, bo, (void*)out, MTOT, DD, DD, 1.0f);
}
